// EmformerAttention_17111149707921
// MI455X (gfx1250) — matrix operand order repaired and checked by reading, not compiled, not run on hardware
//
#include <hip/hip_runtime.h>

// ---- problem constants ----
#define U_LEN   1024
#define B_SZ    8
#define D_MODEL 512
#define NH      8
#define HDIM    64
#define R_LEN   256
#define Q_LEN   1280   // R + U
#define KV_LEN  1280
#define BHEADS  64     // B * NH
#define PE_RAW  2055
#define PE_PAD  2112   // padded so BD p-tiles never read OOB
#define TDIM    1032   // U + RCL  (diagonal band width after rel_shift)
#define NEG_INF -100000000.0f

typedef __attribute__((ext_vector_type(16))) _Float16 v16h;
typedef __attribute__((ext_vector_type(8)))  float    v8f;

union FragU { v16h v; uint4 q[2]; };

// ---- fragment loaders (NT GEMM, row-major sources, contraction contiguous) ----
// A (16x32 f16): lane<16 -> row=lane, K = {k0..k0+7, k0+16..k0+23}
//                lane>=16 -> row=lane-16, K = {k0+8..15, k0+24..31}
__device__ __forceinline__ v16h load_fragA(const _Float16* base, long ld, int row0, int k0) {
  int lane = threadIdx.x & 31;
  const _Float16* p = base + (long)(row0 + (lane & 15)) * ld + k0 + ((lane >> 4) << 3);
  FragU f;
  f.q[0] = *(const uint4*)(p);
  f.q[1] = *(const uint4*)(p + 16);
  return f.v;
}
// B (32x16 f16): lane<16 -> col=lane, K = {k0..k0+7, k0+16..k0+23} ; lane>=16 -> col=lane-16, K = {k0+8..15, k0+24..31}
__device__ __forceinline__ v16h load_fragB(const _Float16* base, long ld, int col0, int k0) {
  int lane = threadIdx.x & 31;
  const _Float16* p = base + (long)(col0 + (lane & 15)) * ld + k0 + ((lane >> 4) << 3);
  FragU f;
  f.q[0] = *(const uint4*)(p);
  f.q[1] = *(const uint4*)(p + 16);
  return f.v;
}

__device__ __forceinline__ v8f wmma16(v16h a, v16h b, v8f c) {
  return __builtin_amdgcn_wmma_f32_16x16x32_f16(false, a, false, b, (short)0, c, false, false);
}

// ---- 64x64 output tile per wave: 4 A-frags x 4 B-frags -> 16 WMMAs per 32-deep K-step ----
__device__ __forceinline__ void mm_64x64(const _Float16* A, long lda, int row0,
                                         const _Float16* Bm, long ldb, int col0,
                                         int K, v8f acc[4][4]) {
  for (int k = 0; k < K; k += 32) {
    v16h a[4], b[4];
#pragma unroll
    for (int j = 0; j < 4; ++j) a[j] = load_fragA(A, lda, row0 + 16 * j, k);
#pragma unroll
    for (int i = 0; i < 4; ++i) b[i] = load_fragB(Bm, ldb, col0 + 16 * i, k);
#pragma unroll
    for (int j = 0; j < 4; ++j)
#pragma unroll
      for (int i = 0; i < 4; ++i)
        acc[j][i] = wmma16(a[j], b[i], acc[j][i]);
  }
}

// ---- 16x64 tile (used by the diagonal-band BD kernel) ----
__device__ __forceinline__ void mm_16x64(const _Float16* A, long lda, int row0,
                                         const _Float16* Bm, long ldb, int col0,
                                         int K, v8f acc[4]) {
  for (int k = 0; k < K; k += 32) {
    v16h a = load_fragA(A, lda, row0, k);
#pragma unroll
    for (int i = 0; i < 4; ++i) {
      v16h b = load_fragB(Bm, ldb, col0 + 16 * i, k);
      acc[i] = wmma16(a, b, acc[i]);
    }
  }
}

// ---- conversion / fill ----
__global__ void cvt_kernel(const float* __restrict__ src, _Float16* __restrict__ dst, int n) {
  int i = blockIdx.x * blockDim.x + threadIdx.x;
  if (i < n) dst[i] = (_Float16)src[i];
}
__global__ void zero_kernel(_Float16* dst, int n) {
  int i = blockIdx.x * blockDim.x + threadIdx.x;
  if (i < n) dst[i] = (_Float16)0.0f;
}

// ---- Q projection: query = X @ Wq.T + bq ; epilogue builds q_u (all Q) and q_v (utterance) heads ----
__global__ __launch_bounds__(32) void gemm_q_kernel(
    const _Float16* __restrict__ Xh, const _Float16* __restrict__ Wh,
    const float* __restrict__ bq, const float* __restrict__ pbu, const float* __restrict__ pbv,
    const float* __restrict__ pbu_s, const float* __restrict__ pbv_s,
    _Float16* __restrict__ qu, _Float16* __restrict__ qv) {
  int row0 = blockIdx.x * 64, col0 = blockIdx.y * 64;
  v8f z = {0,0,0,0,0,0,0,0};
  v8f acc[4][4] = {{z,z,z,z},{z,z,z,z},{z,z,z,z},{z,z,z,z}};
  mm_64x64(Xh, D_MODEL, row0, Wh, D_MODEL, col0, D_MODEL, acc);
  int lane = threadIdx.x & 31, n = lane & 15, mh = (lane >> 4) << 3;
  const float scaling = 0.125f; // HDIM^-0.5
  float eu = __expf(pbu_s[0]), ev = __expf(pbv_s[0]);
#pragma unroll
  for (int j = 0; j < 4; ++j) {
#pragma unroll
    for (int i = 0; i < 4; ++i) {
      int o = col0 + i * 16 + n;
      int h = o >> 6, d = o & 63;
#pragma unroll
      for (int r = 0; r < 8; ++r) {
        int row = row0 + 16 * j + r + mh;   // row = q*B + b
        int q = row >> 3, b = row & 7;
        int bh = b * NH + h;
        float val = acc[j][i][r] + bq[o];
        qu[((size_t)bh * Q_LEN + q) * HDIM + d] = (_Float16)(val * scaling + pbu[o] * eu);
        if (q >= R_LEN)
          qv[((size_t)bh * U_LEN + (q - R_LEN)) * HDIM + d] = (_Float16)(val * scaling + pbv[o] * ev);
      }
    }
  }
}

// ---- KV projection: kv = X @ Wkv.T + bkv ; K row-major heads, V transposed (d, kpos) ----
__global__ __launch_bounds__(32) void gemm_kv_kernel(
    const _Float16* __restrict__ Xh, const _Float16* __restrict__ Wh,
    const float* __restrict__ bkv,
    _Float16* __restrict__ kh, _Float16* __restrict__ vT) {
  int row0 = blockIdx.x * 64, col0 = blockIdx.y * 64;
  v8f z = {0,0,0,0,0,0,0,0};
  v8f acc[4][4] = {{z,z,z,z},{z,z,z,z},{z,z,z,z},{z,z,z,z}};
  mm_64x64(Xh, D_MODEL, row0, Wh, D_MODEL, col0, D_MODEL, acc);
  int lane = threadIdx.x & 31, n = lane & 15, mh = (lane >> 4) << 3;
#pragma unroll
  for (int j = 0; j < 4; ++j) {
#pragma unroll
    for (int i = 0; i < 4; ++i) {
      int o = col0 + i * 16 + n;
#pragma unroll
      for (int r = 0; r < 8; ++r) {
        int row = row0 + 16 * j + r + mh;
        int kp = row >> 3, b = row & 7;
        float val = acc[j][i][r] + bkv[o];
        if (o < D_MODEL) {
          int h = o >> 6, d = o & 63;
          kh[((size_t)(b * NH + h) * KV_LEN + kp) * HDIM + d] = (_Float16)val;
        } else {
          int o2 = o - D_MODEL;
          int h = o2 >> 6, d = o2 & 63;
          vT[(((size_t)(b * NH + h)) * HDIM + d) * KV_LEN + kp] = (_Float16)val;
        }
      }
    }
  }
}

// ---- pos projection: pos = pos_emb @ Wpos.T -> head layout (h, p, d), p padded to PE_PAD ----
__global__ __launch_bounds__(32) void gemm_pos_kernel(
    const _Float16* __restrict__ Ph, const _Float16* __restrict__ Wh,
    _Float16* __restrict__ posh) {
  int row0 = blockIdx.x * 64, col0 = blockIdx.y * 64;
  v8f z = {0,0,0,0,0,0,0,0};
  v8f acc[4][4] = {{z,z,z,z},{z,z,z,z},{z,z,z,z},{z,z,z,z}};
  mm_64x64(Ph, D_MODEL, row0, Wh, D_MODEL, col0, D_MODEL, acc);
  int lane = threadIdx.x & 31, n = lane & 15, mh = (lane >> 4) << 3;
#pragma unroll
  for (int j = 0; j < 4; ++j) {
#pragma unroll
    for (int i = 0; i < 4; ++i) {
      int o = col0 + i * 16 + n;
      int h = o >> 6, d = o & 63;
#pragma unroll
      for (int r = 0; r < 8; ++r) {
        int p = row0 + 16 * j + r + mh;
        posh[((size_t)h * PE_PAD + p) * HDIM + d] = (_Float16)acc[j][i][r];
      }
    }
  }
}

// ---- BD band: mbd[u,p] = q_v[u]·pos[p]; scatter diagonal band t = p-(U-1-u) into (bh,u,t) ----
__global__ __launch_bounds__(32) void bd_kernel(
    const _Float16* __restrict__ qv, const _Float16* __restrict__ posh,
    _Float16* __restrict__ mbdD) {
  int u0 = blockIdx.x * 16;
  int bh = blockIdx.y;
  int h = bh & 7;
  int p0 = (U_LEN - 16 - u0) + (int)blockIdx.z * 64;   // window start is tile-aligned, >= 0
  const _Float16* A  = qv + (size_t)bh * U_LEN * HDIM;
  const _Float16* Bm = posh + (size_t)h * PE_PAD * HDIM;
  v8f z = {0,0,0,0,0,0,0,0};
  v8f acc[4] = {z, z, z, z};
  mm_16x64(A, HDIM, u0, Bm, HDIM, p0, HDIM, acc);
  int lane = threadIdx.x & 31, n = lane & 15, mh = (lane >> 4) << 3;
  _Float16* out = mbdD + (size_t)bh * U_LEN * TDIM;
#pragma unroll
  for (int i = 0; i < 4; ++i) {
#pragma unroll
    for (int r = 0; r < 8; ++r) {
      int u = u0 + r + mh;
      int p = p0 + i * 16 + n;
      int t = p - (U_LEN - 1) + u;  // shifted column index
      if (t >= 0 && t < TDIM)
        out[(size_t)u * TDIM + t] = (_Float16)acc[i][r];
    }
  }
}

// rel-shifted column mapping for key index k (utterance queries only)
__device__ __forceinline__ int jmap(int k) {
  if (k < R_LEN) {
    int ch = k >> 3, rm = k & 7;
    return (ch == 31) ? (U_LEN + rm) : ((ch + 1) * 32 + rm);
  }
  return k - R_LEN;
}

// ---- flash attention: one wave per (bh, 16-row q tile); streams keys 32 at a time ----
__global__ __launch_bounds__(32) void flash_kernel(
    const _Float16* __restrict__ qu, const _Float16* __restrict__ kh,
    const _Float16* __restrict__ vT, const _Float16* __restrict__ mbdD,
    const unsigned char* __restrict__ amask, const int* __restrict__ lengths,
    _Float16* __restrict__ X2) {
  __shared__ __align__(16) _Float16 Plds[16 * 32];
  int q0 = blockIdx.x * 16;
  int bh = blockIdx.y;
  int b = bh >> 3, h = bh & 7;
  int lane = threadIdx.x & 31, n = lane & 15, mh = (lane >> 4) << 3;

  const _Float16* quB = qu   + (size_t)bh * Q_LEN * HDIM;
  const _Float16* kB  = kh   + (size_t)bh * KV_LEN * HDIM;
  const _Float16* vB  = vT   + (size_t)bh * HDIM * KV_LEN;
  const _Float16* bdB = mbdD + (size_t)bh * U_LEN * TDIM;

  v16h aq0 = load_fragA(quB, HDIM, q0, 0);
  v16h aq1 = load_fragA(quB, HDIM, q0, 32);

  v8f z = {0,0,0,0,0,0,0,0};
  v8f O[4] = {z, z, z, z};
  float mrow[8], lrow[8];
#pragma unroll
  for (int r = 0; r < 8; ++r) { mrow[r] = -3.0e38f; lrow[r] = 0.0f; }

  int klim = R_LEN + lengths[b];   // keys >= klim are length-masked

  for (int kk = 0; kk < KV_LEN; kk += 32) {
    v8f s[2] = {z, z};
#pragma unroll
    for (int t = 0; t < 2; ++t) {
      v16h b0 = load_fragB(kB, HDIM, kk + 16 * t, 0);
      v16h b1 = load_fragB(kB, HDIM, kk + 16 * t, 32);
      s[t] = wmma16(aq0, b0, s[t]);
      s[t] = wmma16(aq1, b1, s[t]);
    }
#pragma unroll
    for (int r = 0; r < 8; ++r) {
      int m = r + mh;
      int qrow = q0 + m;
      float tmax = -3.0e38f;
#pragma unroll
      for (int t = 0; t < 2; ++t) {
        int kcol = kk + 16 * t + n;
        float sv = s[t][r];
        if (qrow >= R_LEN) {
          int u = qrow - R_LEN;
          sv += (float)bdB[(size_t)u * TDIM + jmap(kcol)];
        }
        if (amask[(size_t)qrow * KV_LEN + kcol]) sv = NEG_INF;
        if (kcol >= klim) sv = NEG_INF;
        s[t][r] = sv;
        tmax = fmaxf(tmax, sv);
      }
      for (int off = 1; off < 16; off <<= 1) tmax = fmaxf(tmax, __shfl_xor(tmax, off));
      float newm = fmaxf(mrow[r], tmax);
      float corr = __expf(mrow[r] - newm);
      mrow[r] = newm;
      float psum = 0.0f;
#pragma unroll
      for (int t = 0; t < 2; ++t) {
        float p = __expf(s[t][r] - newm);
        psum += p;
        Plds[m * 32 + 16 * t + n] = (_Float16)p;
      }
      for (int off = 1; off < 16; off <<= 1) psum += __shfl_xor(psum, off);
      lrow[r] = lrow[r] * corr + psum;
#pragma unroll
      for (int i = 0; i < 4; ++i) O[i][r] *= corr;
    }
    asm volatile("s_wait_dscnt 0x0" ::: "memory");  // LDS stores visible to this wave's loads
    // P tile (16x32) as A fragment from LDS
    const _Float16* pp = &Plds[(lane & 15) * 32 + ((lane >> 4) << 3)];
    FragU pf;
    pf.q[0] = *(const uint4*)(pp);
    pf.q[1] = *(const uint4*)(pp + 16);
#pragma unroll
    for (int i = 0; i < 4; ++i) {
      v16h bv = load_fragB(vB, KV_LEN, i * 16, kk);
      O[i] = wmma16(pf.v, bv, O[i]);
    }
    asm volatile("s_wait_dscnt 0x0" ::: "memory");  // loads done before next-iter overwrite
  }
#pragma unroll
  for (int r = 0; r < 8; ++r) {
    int qrow = q0 + r + mh;
    float inv = 1.0f / lrow[r];
#pragma unroll
    for (int i = 0; i < 4; ++i) {
      int d = i * 16 + n;
      X2[((size_t)qrow * B_SZ + b) * D_MODEL + h * HDIM + d] = (_Float16)(O[i][r] * inv);
    }
  }
}

// ---- output projection: out = X2 @ Wo.T + bo (f32 output) ----
__global__ __launch_bounds__(32) void gemm_o_kernel(
    const _Float16* __restrict__ Xh, const _Float16* __restrict__ Wh,
    const float* __restrict__ bo, float* __restrict__ out) {
  int row0 = blockIdx.x * 64, col0 = blockIdx.y * 64;
  v8f z = {0,0,0,0,0,0,0,0};
  v8f acc[4][4] = {{z,z,z,z},{z,z,z,z},{z,z,z,z},{z,z,z,z}};
  mm_64x64(Xh, D_MODEL, row0, Wh, D_MODEL, col0, D_MODEL, acc);
  int lane = threadIdx.x & 31, n = lane & 15, mh = (lane >> 4) << 3;
#pragma unroll
  for (int j = 0; j < 4; ++j) {
#pragma unroll
    for (int i = 0; i < 4; ++i) {
      int o = col0 + i * 16 + n;
#pragma unroll
      for (int r = 0; r < 8; ++r) {
        int row = row0 + 16 * j + r + mh;
        out[(size_t)row * D_MODEL + o] = acc[j][i][r] + bo[o];
      }
    }
  }
}

extern "C" void kernel_launch(void* const* d_in, const int* in_sizes, int n_in,
                              void* d_out, int out_size, void* d_ws, size_t ws_size,
                              hipStream_t stream) {
  (void)in_sizes; (void)n_in; (void)out_size; (void)ws_size;
  const float* utt     = (const float*)d_in[0];
  const int*   lengths = (const int*)d_in[1];
  const float* rc      = (const float*)d_in[2];
  // d_in[3] summary (empty), d_in[4] memory (empty)
  const unsigned char* amask = (const unsigned char*)d_in[5];
  const float* pos_emb = (const float*)d_in[6];
  const float* Wq   = (const float*)d_in[7];
  const float* bq   = (const float*)d_in[8];
  const float* Wkv  = (const float*)d_in[9];
  const float* bkv  = (const float*)d_in[10];
  const float* Wo   = (const float*)d_in[11];
  const float* bo   = (const float*)d_in[12];
  const float* Wpos = (const float*)d_in[13];
  const float* pbu  = (const float*)d_in[14];
  const float* pbv  = (const float*)d_in[15];
  const float* pbu_s = (const float*)d_in[16];
  const float* pbv_s = (const float*)d_in[17];
  float* out = (float*)d_out;

  char* ws = (char*)d_ws;
  size_t off = 0;
  auto alloc = [&](size_t elems) {
    _Float16* p = (_Float16*)(ws + off);
    off += ((elems * sizeof(_Float16) + 255) / 256) * 256;
    return p;
  };
  const int NROW = Q_LEN * B_SZ;                 // 10240
  _Float16* Xh      = alloc((size_t)NROW * D_MODEL);
  _Float16* Wq_h    = alloc((size_t)D_MODEL * D_MODEL);
  _Float16* Wkv_h   = alloc((size_t)2 * D_MODEL * D_MODEL);
  _Float16* Wpos_h  = alloc((size_t)D_MODEL * D_MODEL);
  _Float16* Wo_h    = alloc((size_t)D_MODEL * D_MODEL);
  _Float16* posem_h = alloc((size_t)PE_PAD * D_MODEL);
  _Float16* pos_h   = alloc((size_t)NH * PE_PAD * HDIM);
  _Float16* qu_h    = alloc((size_t)BHEADS * Q_LEN * HDIM);
  _Float16* qv_h    = alloc((size_t)BHEADS * U_LEN * HDIM);
  _Float16* k_h     = alloc((size_t)BHEADS * KV_LEN * HDIM);
  _Float16* vT_h    = alloc((size_t)BHEADS * HDIM * KV_LEN);
  _Float16* X2_h    = alloc((size_t)NROW * D_MODEL);
  _Float16* mbdD_h  = alloc((size_t)BHEADS * U_LEN * TDIM);

  auto cvt = [&](const float* s, _Float16* d, int n) {
    cvt_kernel<<<(n + 255) / 256, 256, 0, stream>>>(s, d, n);
  };
  cvt(rc,  Xh,                          R_LEN * B_SZ * D_MODEL);
  cvt(utt, Xh + R_LEN * B_SZ * D_MODEL, U_LEN * B_SZ * D_MODEL);
  cvt(Wq,   Wq_h,   D_MODEL * D_MODEL);
  cvt(Wkv,  Wkv_h,  2 * D_MODEL * D_MODEL);
  cvt(Wpos, Wpos_h, D_MODEL * D_MODEL);
  cvt(Wo,   Wo_h,   D_MODEL * D_MODEL);
  cvt(pos_emb, posem_h, PE_RAW * D_MODEL);
  {
    int npad = (PE_PAD - PE_RAW) * D_MODEL;
    zero_kernel<<<(npad + 255) / 256, 256, 0, stream>>>(posem_h + PE_RAW * D_MODEL, npad);
  }

  gemm_q_kernel<<<dim3(NROW / 64, D_MODEL / 64), 32, 0, stream>>>(
      Xh, Wq_h, bq, pbu, pbv, pbu_s, pbv_s, qu_h, qv_h);
  gemm_kv_kernel<<<dim3(NROW / 64, (2 * D_MODEL) / 64), 32, 0, stream>>>(
      Xh, Wkv_h, bkv, k_h, vT_h);
  gemm_pos_kernel<<<dim3(PE_PAD / 64, D_MODEL / 64), 32, 0, stream>>>(
      posem_h, Wpos_h, pos_h);
  bd_kernel<<<dim3(U_LEN / 16, BHEADS, 17), 32, 0, stream>>>(qv_h, pos_h, mbdD_h);
  flash_kernel<<<dim3(Q_LEN / 16, BHEADS), 32, 0, stream>>>(
      qu_h, k_h, vT_h, mbdD_h, amask, lengths, X2_h);
  gemm_o_kernel<<<dim3(NROW / 64, D_MODEL / 64), 32, 0, stream>>>(X2_h, Wo_h, bo, out);
}
